// Convolution_19851338842926
// MI455X (gfx1250) — matrix operand order repaired and checked by reading, not compiled, not run on hardware
//
#include <hip/hip_runtime.h>
#include <hip/hip_bf16.h>
#include <stdint.h>

#define NPTS 131072
#define NK   125

typedef __attribute__((ext_vector_type(16))) __bf16 v16bf;
typedef __attribute__((ext_vector_type(8)))  __bf16 v8bf;
typedef __attribute__((ext_vector_type(8)))  float  v8f;

__device__ __forceinline__ unsigned short f2bf(float f) {
  unsigned u = __float_as_uint(f);
  u += 0x7FFFu + ((u >> 16) & 1u);   // round-to-nearest-even
  return (unsigned short)(u >> 16);
}

// ---------------------------------------------------------------------------
// Kernel 1: x_feat f32 -> bf16, with a zero pad row at index NPTS
// ---------------------------------------------------------------------------
__global__ void cvt_x_kernel(const float* __restrict__ x,
                             unsigned short* __restrict__ xb,
                             int total, int nvalid) {
  int e = blockIdx.x * 256 + threadIdx.x;
  if (e < total) xb[e] = (e < nvalid) ? f2bf(x[e]) : (unsigned short)0;
}

// ---------------------------------------------------------------------------
// Kernel 2: build (125,128,128) equivariant kernel, fold self-connection into
// kern[62], store bf16 pre-packed as WMMA B-fragments:
//   frag (k, ktile, ntile): 32 lanes x 16 bf16
//   B 32x16 layout: lane = n + 16*((kk/8)%2), slot = kk%8 + 8*(kk>=16)
// ---------------------------------------------------------------------------
__global__ void build_kern_kernel(const float* __restrict__ weight,
                                  const float* __restrict__ wss,
                                  const float* __restrict__ wsv,
                                  unsigned short* __restrict__ kp) {
  __shared__ float W[4096];
  const int k = blockIdx.x, tid = threadIdx.x;
  const int zi = k / 25, yi = (k / 5) % 5, xi = k % 5;
  const float px = (float)(xi - 2), py = (float)(yi - 2), pz = (float)(zi - 2);
  const float norm = sqrtf(px * px + py * py + pz * pz);
  const float step = 2.5f / 6.0f;
  float emb[5];
#pragma unroll
  for (int b = 0; b < 5; ++b) {
    float d = (norm - (float)(b + 1) * step) / step;
    if (fabsf(d) < 1.0f) {
      float ds = fminf(fmaxf(d, -1.0f + 1e-6f), 1.0f - 1e-6f);
      emb[b] = 1.14136f * __expf(2.0f - 1.0f / (1.0f + ds) - 1.0f / (1.0f - ds));
    } else emb[b] = 0.0f;
  }
  const float inorm = (norm > 0.0f) ? 1.0f / fmaxf(norm, 1e-9f) : 0.0f;
  const float sq3 = 1.7320508075688772f;
  const float Y1[3] = { sq3 * py * inorm, sq3 * pz * inorm, sq3 * px * inorm };

  for (int j = tid; j < 4096; j += 256) {
    float s = 0.0f;
#pragma unroll
    for (int b = 0; b < 5; ++b) s += emb[b] * weight[b * 4096 + j];
    W[j] = s * (1.0f / 125.0f);
  }
  __syncthreads();

  const float c = 0.125f;                 // 1/sqrt(64)
  const float cd = c / sq3;
  const float inv = 0.17677669529663687f; // 1/sqrt(32)

  for (int e = tid; e < 16384; e += 256) {
    const int i = e >> 7, j = e & 127;
    float val;
    if (i < 32) {
      if (j < 32) {
        val = c * W[i * 32 + j];                       // Kss
      } else {
        int jj = j - 32, w = jj / 3, kk = jj - 3 * w;
        val = c * W[1024 + i * 32 + w] * Y1[kk];       // Ksv (Wb)
      }
    } else {
      int ii = i - 32, u = ii / 3, ic = ii - 3 * u;
      if (j < 32) {
        val = cd * W[3072 + u * 32 + j] * Y1[ic];      // Kvs (Wd)
      } else {
        int jj = j - 32, w = jj / 3, kk = jj - 3 * w;
        val = (ic == kk) ? c * W[2048 + u * 32 + w] : 0.0f; // Kvv (Wc)
      }
    }
    if (k == 62) {  // fold self-connection into the (0,0,0) offset
      if (i < 32 && j < 32) {
        val += wss[i * 32 + j] * inv;
      } else if (i >= 32 && j >= 32) {
        int ii = i - 32, u = ii / 3, ic = ii - 3 * u;
        int jj = j - 32, w = jj / 3, kk = jj - 3 * w;
        if (ic == kk) val += wsv[u * 32 + w] * inv;
      }
    }
    // pack into B-fragment layout
    const int ktile = i >> 5, kin = i & 31;
    const int ntile = j >> 4;
    const int lane = (j & 15) | (((kin >> 3) & 1) << 4);
    const int slot = (kin & 7) | ((kin >> 4) << 3);
    const size_t off = ((size_t)(k * 32 + ktile * 8 + ntile) << 9) + (size_t)lane * 16 + slot;
    kp[off] = f2bf(val);
  }
}

// ---------------------------------------------------------------------------
// Kernel 3: main gathered GEMM.  512 blocks x 256 threads (8 waves).
// Each wave owns TWO 16-row M-tiles (M=32) so every B fragment pulled from
// LDS feeds two WMMAs from registers (halves LDS bytes per WMMA).
// Double-buffered global_load_async_to_lds_b128 stages kern[k] per block.
// ---------------------------------------------------------------------------
__global__ void __launch_bounds__(256)
conv_wmma_kernel(const unsigned short* __restrict__ xb,
                 const unsigned short* __restrict__ kp,
                 const int* __restrict__ nbr,
                 float* __restrict__ out) {
  __shared__ unsigned short ldsK[2][16384];   // 2 x 32 KB
  const int tid  = threadIdx.x;
  const int lane = tid & 31;
  const int wave = tid >> 5;
  const int h    = lane >> 4;
  const int col  = lane & 15;
  const int row0 = blockIdx.x * 256 + wave * 32 + col;  // M-tile 0
  const int row1 = row0 + 16;                           // M-tile 1

  const unsigned ldsBase0 = (unsigned)(uintptr_t)(&ldsK[0][0]);
  const unsigned ldsBase1 = (unsigned)(uintptr_t)(&ldsK[1][0]);
  const unsigned voffTid  = (unsigned)tid * 16u;

  // prologue: async-stage kern[0] into buffer 0 (each thread: 8 x 16B)
  {
    const uint64_t src = (uint64_t)(uintptr_t)kp;
#pragma unroll
    for (int i = 0; i < 8; ++i) {
      unsigned l = ldsBase0 + voffTid + (unsigned)i * 4096u;
      unsigned v = voffTid + (unsigned)i * 4096u;
      asm volatile("global_load_async_to_lds_b128 %0, %1, %2"
                   :: "v"(l), "v"(v), "s"(src) : "memory");
    }
  }

  v8f acc0[8] = {};
  v8f acc1[8] = {};

  for (int k = 0; k < NK; ++k) {
    if (k + 1 < NK) {
      const uint64_t src = (uint64_t)(uintptr_t)(kp + (size_t)(k + 1) * 16384);
      const unsigned base = ((k + 1) & 1) ? ldsBase1 : ldsBase0;
#pragma unroll
      for (int i = 0; i < 8; ++i) {
        unsigned l = base + voffTid + (unsigned)i * 4096u;
        unsigned v = voffTid + (unsigned)i * 4096u;
        asm volatile("global_load_async_to_lds_b128 %0, %1, %2"
                     :: "v"(l), "v"(v), "s"(src) : "memory");
      }
      asm volatile("s_wait_asynccnt 8" ::: "memory");   // kern[k] staged
    } else {
      asm volatile("s_wait_asynccnt 0" ::: "memory");
    }
    __syncthreads();

    // gather A rows (bf16); 16-bit A 16x32 layout:
    // lane half h: elems 0..7 = K 8h..8h+7, elems 8..15 = K 16+8h..16+8h+7
    const int idx0 = nbr[(size_t)k * NPTS + row0];
    const int idx1 = nbr[(size_t)k * NPTS + row1];
    const unsigned short* arow0 = xb + (size_t)idx0 * 128 + 8 * h;
    const unsigned short* arow1 = xb + (size_t)idx1 * 128 + 8 * h;
    const unsigned short* lb = &ldsK[k & 1][0] + lane * 16;

#pragma unroll
    for (int c = 0; c < 4; ++c) {
      v8bf lo0 = *(const v8bf*)(arow0 + 32 * c);
      v8bf hi0 = *(const v8bf*)(arow0 + 32 * c + 16);
      v8bf lo1 = *(const v8bf*)(arow1 + 32 * c);
      v8bf hi1 = *(const v8bf*)(arow1 + 32 * c + 16);
      v16bf a0 = __builtin_shufflevector(lo0, hi0, 0, 1, 2, 3, 4, 5, 6, 7,
                                         8, 9, 10, 11, 12, 13, 14, 15);
      v16bf a1 = __builtin_shufflevector(lo1, hi1, 0, 1, 2, 3, 4, 5, 6, 7,
                                         8, 9, 10, 11, 12, 13, 14, 15);
#pragma unroll
      for (int nt = 0; nt < 8; ++nt) {
        v16bf b = *(const v16bf*)(lb + (size_t)(c * 8 + nt) * 512);
        acc0[nt] = __builtin_amdgcn_wmma_f32_16x16x32_bf16(
            false, a0, false, b, (short)0, acc0[nt], false, false);
        acc1[nt] = __builtin_amdgcn_wmma_f32_16x16x32_bf16(
            false, a1, false, b, (short)0, acc1[nt], false, false);
      }
    }
    __syncthreads();   // all waves done reading buf[k&1] before it is refilled
  }

  // C/D layout: VGPR g, lanes 0-15: M=g, lanes 16-31: M=8+g; N = lane%16
  const int rbase0 = blockIdx.x * 256 + wave * 32 + 8 * h;
  const int rbase1 = rbase0 + 16;
#pragma unroll
  for (int nt = 0; nt < 8; ++nt) {
#pragma unroll
    for (int g = 0; g < 8; ++g) {
      out[(size_t)(rbase0 + g) * 128 + nt * 16 + col] = acc0[nt][g];
      out[(size_t)(rbase1 + g) * 128 + nt * 16 + col] = acc1[nt][g];
    }
  }
}

// ---------------------------------------------------------------------------
extern "C" void kernel_launch(void* const* d_in, const int* in_sizes, int n_in,
                              void* d_out, int out_size, void* d_ws, size_t ws_size,
                              hipStream_t stream) {
  const float* x_feat = (const float*)d_in[0];      // N x 128
  const float* weight = (const float*)d_in[1];      // 5 x 4096
  const float* wss    = (const float*)d_in[2];      // 32 x 32
  const float* wsv    = (const float*)d_in[3];      // 32 x 32
  const int*   nbr    = (const int*)d_in[4];        // 125 x N

  char* ws = (char*)d_ws;
  unsigned short* xb = (unsigned short*)ws;                                  // (N+1)*128 bf16
  unsigned short* kp = (unsigned short*)(ws + (size_t)(NPTS + 1) * 128 * 2); // 125*16384 bf16
  float* outp = (float*)d_out;

  const int total  = (NPTS + 1) * 128;
  const int nvalid = NPTS * 128;
  cvt_x_kernel<<<(total + 255) / 256, 256, 0, stream>>>(x_feat, xb, total, nvalid);
  build_kern_kernel<<<NK, 256, 0, stream>>>(weight, wss, wsv, kp);
  conv_wmma_kernel<<<NPTS / 256, 256, 0, stream>>>(xb, kp, nbr, outp);
  (void)in_sizes; (void)n_in; (void)out_size; (void)ws_size;
}
